// HybridKernelLSTM_65481071404582
// MI455X (gfx1250) — matrix operand order repaired and checked by reading, not compiled, not run on hardware
//
#include <hip/hip_runtime.h>
#include <hip/hip_bf16.h>
#include <math.h>

// Quantum-LSTM tagger for gfx1250 (MI455X, wave32, WMMA).
//
// Decomposition:
//  1) qlstm_prep : pack Wcat=(Wf|Wi|Wu|Wo)[0:512] into f16 WMMA B fragments + bcat.
//  2) qlstm_gemm : pre[t, 0:16] = emb[sentence[t]] @ Wcat + bcat  via
//                  v_wmma_f32_16x16x32_f16 (gather fused into A-fragment loads).
//  3) qlstm_scan : sequential LSTM scan. Quantum layer collapses exactly to
//                  z_i = prod_{w<=i} cos(ang_w)*cos(p_w)  (cumprod of cosines).
//  4) qlstm_head : logits = hs @ W_tag + b_tag, log_softmax; one wave32 per row,
//                  shfl_xor reductions.

#define SEQ  4096
#define EMBD 512
#define NQ   4
#define TAGS 64

typedef __attribute__((ext_vector_type(16))) _Float16 v16h;
typedef __attribute__((ext_vector_type(8)))  float    v8f;

// ---------------------------------------------------------------------------
// Kernel 1: pack B fragments + bias vector.
// packedB[kstep(16)][lane(32)][elem(16)] halfs:
//   col n = lane&15 ; K = kstep*32 + (lane<16 ? 0 : 8) + (elem<8 ? elem : elem+8)
__global__ void qlstm_prep(const float* __restrict__ Wf, const float* __restrict__ Wi,
                           const float* __restrict__ Wu, const float* __restrict__ Wo,
                           const float* __restrict__ bf, const float* __restrict__ bi,
                           const float* __restrict__ bu, const float* __restrict__ bo,
                           _Float16* __restrict__ packedB, float* __restrict__ bcat) {
  int idx = blockIdx.x * blockDim.x + threadIdx.x;
  if (idx < 16) {
    int g = idx >> 2, w = idx & 3;
    const float* bp = (g == 0) ? bf : (g == 1) ? bi : (g == 2) ? bu : bo;
    bcat[idx] = bp[w];
  }
  if (idx >= 16 * 32 * 16) return;
  int e     = idx & 15;
  int lane  = (idx >> 4) & 31;
  int kstep = idx >> 9;
  int n = lane & 15;
  int g = n >> 2, w = n & 3;
  int k = kstep * 32 + ((lane < 16) ? 0 : 8) + ((e < 8) ? e : e + 8);   // 0..511
  const float* Wp = (g == 0) ? Wf : (g == 1) ? Wi : (g == 2) ? Wu : Wo;
  packedB[idx] = (_Float16)Wp[k * NQ + w];
}

// ---------------------------------------------------------------------------
// Kernel 2: gather-GEMM, one wave per 16-timestep tile.
// A fragment (16-bit A 16x32): lane L in [0,16): row L, K in {kk+0..7, kk+16..23}
//                              lane L+16    : row L, K in {kk+8..15, kk+24..31}
// C/D layout: VGPR r: lanes 0-15 -> (M=r, N=lane); lanes 16-31 -> (M=8+r, N=lane-16)
__global__ void __launch_bounds__(32)
qlstm_gemm(const int* __restrict__ sentence, const float* __restrict__ emb,
           const _Float16* __restrict__ packedB, const float* __restrict__ bcat,
           float* __restrict__ pre) {
  const int lane = threadIdx.x;
  const int tile = blockIdx.x;           // 256 tiles of 16 timesteps
  const int m    = lane & 15;
  const int s    = (lane < 16) ? 0 : 8;
  const int tok  = sentence[tile * 16 + m];
  const float* row = emb + (size_t)tok * EMBD;

  v8f c = {};
#pragma unroll 4
  for (int kk = 0; kk < EMBD; kk += 32) {
    const float4* p0 = (const float4*)(row + kk + s);
    const float4* p1 = (const float4*)(row + kk + 16 + s);
    float4 f0 = p0[0], f1 = p0[1];
    float4 f2 = p1[0], f3 = p1[1];
    v16h a;
    a[0]  = (_Float16)f0.x; a[1]  = (_Float16)f0.y;
    a[2]  = (_Float16)f0.z; a[3]  = (_Float16)f0.w;
    a[4]  = (_Float16)f1.x; a[5]  = (_Float16)f1.y;
    a[6]  = (_Float16)f1.z; a[7]  = (_Float16)f1.w;
    a[8]  = (_Float16)f2.x; a[9]  = (_Float16)f2.y;
    a[10] = (_Float16)f2.z; a[11] = (_Float16)f2.w;
    a[12] = (_Float16)f3.x; a[13] = (_Float16)f3.y;
    a[14] = (_Float16)f3.z; a[15] = (_Float16)f3.w;
    v16h b = *(const v16h*)(packedB + (size_t)(kk >> 5) * 512 + lane * 16);
    c = __builtin_amdgcn_wmma_f32_16x16x32_f16(false, a, false, b,
                                               (short)0, c, false, false);
  }
  const int n     = lane & 15;
  const float bia = bcat[n];
  const int rbase = tile * 16 + ((lane < 16) ? 0 : 8);
#pragma unroll
  for (int r = 0; r < 8; ++r)
    pre[(size_t)(rbase + r) * 16 + n] = c[r] + bia;
}

// ---------------------------------------------------------------------------
// Kernel 3: sequential scan (latency-bound; single lane).
__global__ void qlstm_scan(const float* __restrict__ pre,
                           const float* __restrict__ Wf, const float* __restrict__ Wi,
                           const float* __restrict__ Wu, const float* __restrict__ Wo,
                           const float* __restrict__ qf, const float* __restrict__ qi,
                           const float* __restrict__ qu, const float* __restrict__ qo,
                           float* __restrict__ hs) {
  if (threadIdx.x != 0 || blockIdx.x != 0) return;
  const float* Wptr[4] = {Wf, Wi, Wu, Wo};
  const float* qptr[4] = {qf, qi, qu, qo};
  float Wh[4][4][4];                      // [gate][j(h-in)][w(out)]
  float cP[4][4];
#pragma unroll
  for (int g = 0; g < 4; ++g) {
#pragma unroll
    for (int w = 0; w < 4; ++w) {
      cP[g][w] = __cosf(qptr[g][w]);
#pragma unroll
      for (int j = 0; j < 4; ++j)
        Wh[g][j][w] = Wptr[g][(EMBD + j) * NQ + w];
    }
  }
  float h[4] = {0.f, 0.f, 0.f, 0.f}, cs[4] = {0.f, 0.f, 0.f, 0.f};
  for (int t = 0; t < SEQ; ++t) {
    const float* pt = pre + (size_t)t * 16;
    float z[4][4];
#pragma unroll
    for (int g = 0; g < 4; ++g) {
      float cum = 1.f;
#pragma unroll
      for (int w = 0; w < 4; ++w) {
        float ang = pt[g * 4 + w]
                  + h[0] * Wh[g][0][w] + h[1] * Wh[g][1][w]
                  + h[2] * Wh[g][2][w] + h[3] * Wh[g][3][w];
        cum *= __cosf(ang) * cP[g][w];
        z[g][w] = cum;
      }
    }
#pragma unroll
    for (int w = 0; w < 4; ++w) {
      float fg = 1.f / (1.f + __expf(-z[0][w]));
      float ig = 1.f / (1.f + __expf(-z[1][w]));
      float e2 = __expf(2.f * z[2][w]);
      float gg = (e2 - 1.f) / (e2 + 1.f);        // tanh
      float og = 1.f / (1.f + __expf(-z[3][w]));
      cs[w] = fg * cs[w] + ig * gg;
      float ec = __expf(2.f * cs[w]);
      h[w] = og * ((ec - 1.f) / (ec + 1.f));
      hs[(size_t)t * 4 + w] = h[w];
    }
  }
}

// ---------------------------------------------------------------------------
// Kernel 4: tag logits + log_softmax. One wave32 per timestep, 2 tags/lane.
__global__ void qlstm_head(const float* __restrict__ hs,
                           const float* __restrict__ Wtag,
                           const float* __restrict__ btag,
                           float* __restrict__ out) {
  const int lane = threadIdx.x & 31;
  const int wave = threadIdx.x >> 5;
  const int t = blockIdx.x * (blockDim.x >> 5) + wave;
  if (t >= SEQ) return;
  const float h0 = hs[t * 4 + 0], h1 = hs[t * 4 + 1];
  const float h2 = hs[t * 4 + 2], h3 = hs[t * 4 + 3];
  const int j0 = lane, j1 = lane + 32;
  float l0 = btag[j0] + h0 * Wtag[0 * TAGS + j0] + h1 * Wtag[1 * TAGS + j0]
                      + h2 * Wtag[2 * TAGS + j0] + h3 * Wtag[3 * TAGS + j0];
  float l1 = btag[j1] + h0 * Wtag[0 * TAGS + j1] + h1 * Wtag[1 * TAGS + j1]
                      + h2 * Wtag[2 * TAGS + j1] + h3 * Wtag[3 * TAGS + j1];
  float m = fmaxf(l0, l1);
#pragma unroll
  for (int off = 16; off > 0; off >>= 1)
    m = fmaxf(m, __shfl_xor(m, off, 32));
  float s = __expf(l0 - m) + __expf(l1 - m);
#pragma unroll
  for (int off = 16; off > 0; off >>= 1)
    s += __shfl_xor(s, off, 32);
  const float lse = m + __logf(s);
  out[(size_t)t * TAGS + j0] = l0 - lse;
  out[(size_t)t * TAGS + j1] = l1 - lse;
}

// ---------------------------------------------------------------------------
extern "C" void kernel_launch(void* const* d_in, const int* in_sizes, int n_in,
                              void* d_out, int out_size, void* d_ws, size_t ws_size,
                              hipStream_t stream) {
  (void)in_sizes; (void)n_in; (void)out_size; (void)ws_size;
  const int*   sentence = (const int*)d_in[0];
  const float* emb  = (const float*)d_in[1];
  const float* Wf   = (const float*)d_in[2];
  const float* bf   = (const float*)d_in[3];
  const float* Wi   = (const float*)d_in[4];
  const float* bi   = (const float*)d_in[5];
  const float* Wu   = (const float*)d_in[6];
  const float* bu   = (const float*)d_in[7];
  const float* Wo   = (const float*)d_in[8];
  const float* bo   = (const float*)d_in[9];
  const float* qf   = (const float*)d_in[10];
  const float* qi   = (const float*)d_in[11];
  const float* qu   = (const float*)d_in[12];
  const float* qo   = (const float*)d_in[13];
  const float* Wtag = (const float*)d_in[14];
  const float* btag = (const float*)d_in[15];
  float* out = (float*)d_out;

  char* ws = (char*)d_ws;
  _Float16* packedB = (_Float16*)(ws);                 // 16 KB
  float*    bcat    = (float*)(ws + 16384);            // 64 B
  float*    pre     = (float*)(ws + 32768);            // 4096*16*4 = 256 KB
  float*    hsbuf   = (float*)(ws + 32768 + 262144);   // 4096*4*4  = 64 KB

  qlstm_prep<<<(16 * 32 * 16 + 255) / 256, 256, 0, stream>>>(
      Wf, Wi, Wu, Wo, bf, bi, bu, bo, packedB, bcat);
  qlstm_gemm<<<SEQ / 16, 32, 0, stream>>>(sentence, emb, packedB, bcat, pre);
  qlstm_scan<<<1, 32, 0, stream>>>(pre, Wf, Wi, Wu, Wo, qf, qi, qu, qo, hsbuf);
  qlstm_head<<<SEQ / 8, 256, 0, stream>>>(hsbuf, Wtag, btag, out);
}
